// CILRSModel_33337536151555
// MI455X (gfx1250) — compile-verified, matrix operand order checked by reading, not run on hardware
//
#include <hip/hip_runtime.h>

// ---------------------------------------------------------------------------
// CILRS forward, MI455X (gfx1250). wave32, WMMA f32<=f16 16x16x32.
// B=65536, D_EMB=512, D_LAT=128, D_IN=640, H=256, K=6 branches + speed head.
// ---------------------------------------------------------------------------

typedef __attribute__((ext_vector_type(16))) _Float16     v16h;
typedef __attribute__((ext_vector_type(8)))  float        v8f;
typedef __attribute__((ext_vector_type(4)))  unsigned int v4u;
typedef __attribute__((ext_vector_type(2)))  unsigned int v2u;

union Frag { v4u u[2]; v16h v; };

__device__ __forceinline__ int perm32(int k) {
  // fragment-order permutation: [K0..7, K16..23, K8..15, K24..31]
  return (k < 8 || k >= 24) ? k : (k < 16 ? k + 8 : k - 8);
}

__device__ __forceinline__ v8f wmma16(const Frag& a, const Frag& b, v8f c) {
  return __builtin_amdgcn_wmma_f32_16x16x32_f16(false, a.v, false, b.v,
                                                (short)0, c, false, false);
}

#define NB   6      // branch heads
#define NH   7      // 6 branches + speed head
#define DIN  640
#define HH   256
#define DLAT 128
#define KS1  20     // 640/32
#define KS2  8      // 256/32
#define MT   64     // rows per block

// --------------------------- weight pre-pack ------------------------------
// W1s: [7][20][256][32] halves, W2s: [7][8][256][32], Wsi2s: [8][128][32]
__global__ __launch_bounds__(256) void cilrs_pack(
    const float* __restrict__ Wb1, const float* __restrict__ Wso1,
    const float* __restrict__ Wb2, const float* __restrict__ Wso2,
    const float* __restrict__ Wsi2,
    _Float16* __restrict__ W1s, _Float16* __restrict__ W2s,
    _Float16* __restrict__ Wsi2s)
{
  int idx = blockIdx.x * 256 + threadIdx.x;
  const int E1 = NH * DIN * HH;      // 1,146,880
  const int E2 = NH * HH * HH;       //   458,752
  const int E3 = HH * DLAT;          //    32,768
  if (idx < E1) {
    int h = idx / (DIN * HH), r = idx % (DIN * HH), k = r >> 8, n = r & 255;
    float s = (h < 6) ? Wb1[idx] : Wso1[r];
    W1s[(((size_t)h * KS1 + (k >> 5)) * HH + n) * 32 + perm32(k & 31)] = (_Float16)s;
  } else if (idx < E1 + E2) {
    int j = idx - E1;
    int h = j / (HH * HH), r = j % (HH * HH), k = r >> 8, n = r & 255;
    float s = (h < 6) ? Wb2[j] : Wso2[r];
    W2s[(((size_t)h * KS2 + (k >> 5)) * HH + n) * 32 + perm32(k & 31)] = (_Float16)s;
  } else if (idx < E1 + E2 + E3) {
    int j = idx - E1 - E2;
    int k = j >> 7, n = j & 127;
    Wsi2s[(((size_t)(k >> 5) * DLAT) + n) * 32 + perm32(k & 31)] = (_Float16)Wsi2[j];
  }
}

// ------------------------------ main kernel -------------------------------
__global__ __launch_bounds__(256) void cilrs_main(
    const float* __restrict__ emb, const float* __restrict__ speed,
    const int*   __restrict__ cmd,
    const float* __restrict__ Wsi1, const float* __restrict__ bsi1,
    const float* __restrict__ bsi2,
    const float* __restrict__ bso1, const float* __restrict__ bso2,
    const float* __restrict__ bso3,
    const float* __restrict__ bb1,  const float* __restrict__ bb2,
    const float* __restrict__ bb3,
    const float* __restrict__ Wso3, const float* __restrict__ Wb3,
    const _Float16* __restrict__ W1s, const _Float16* __restrict__ W2s,
    const _Float16* __restrict__ Wsi2s,
    float* __restrict__ outCtrl, float* __restrict__ outSpd)
{
  __shared__ _Float16 Atile[MT * DIN];   // 80 KB: [emb(512) | sp(128)] per row
  __shared__ _Float16 Xa[MT * HH];       // 32 KB: hsp / layer-1 activations
  __shared__ _Float16 Xb[MT * HH];       // 32 KB: layer-2 activations
  __shared__ float    w3lds[HH * 4];     //  4 KB: layer-3 weights (padded to 4)
  __shared__ float    spd[MT];

  const int tid    = threadIdx.x;
  const int wid    = tid >> 5;
  const int lane   = tid & 31;
  const int laneLo = lane & 15;
  const int hiSel  = lane >> 4;          // 0: lanes 0-15, 1: lanes 16-31
  const int mBase  = blockIdx.x * MT;

  // ---- Phase 0: embedding f32 -> f16 into Atile[:, 0:512] ----
  for (int q = tid; q < MT * 128; q += 256) {
    int row = q >> 7, c4 = (q & 127) << 2;
    float4 v = *(const float4*)(emb + (size_t)(mBase + row) * 512 + c4);
    union { _Float16 h[4]; v2u u; } p;
    p.h[0] = (_Float16)v.x; p.h[1] = (_Float16)v.y;
    p.h[2] = (_Float16)v.z; p.h[3] = (_Float16)v.w;
    *(v2u*)&Atile[row * DIN + c4] = p.u;
  }
  if (tid < MT) spd[tid] = speed[mBase + tid];
  __syncthreads();

  // ---- Phase 0b: hsp = relu(speed * Wsi1 + bsi1) -> Xa [64][256] ----
  for (int q = tid; q < MT * HH; q += 256) {
    int row = q >> 8, j = q & 255;
    float hv = fmaf(spd[row], Wsi1[j], bsi1[j]);
    Xa[q] = (_Float16)(hv > 0.f ? hv : 0.f);
  }
  __syncthreads();

  // ---- Phase 0c: sp = hsp @ Wsi2 + bsi2 -> Atile[:, 512:640] (WMMA) ----
  {
    const int wm2 = wid >> 1, wn2 = wid & 1;     // 4x2 wave grid: 16x64 each
    v8f acc[4];
    #pragma unroll
    for (int cb = 0; cb < 4; ++cb) acc[cb] = (v8f){0,0,0,0,0,0,0,0};
    const int row = wm2 * 16 + laneLo;
    for (int ks = 0; ks < KS2; ++ks) {
      Frag a;
      int kh = ks * 32 + hiSel * 8;
      a.u[0] = *(const v4u*)&Xa[row * HH + kh];
      a.u[1] = *(const v4u*)&Xa[row * HH + kh + 16];
      #pragma unroll
      for (int cb = 0; cb < 4; ++cb) {
        int n = wn2 * 64 + cb * 16 + laneLo;
        const v4u* bp = (const v4u*)&Wsi2s[((size_t)ks * DLAT + n) * 32 + hiSel * 16];
        Frag b; b.u[0] = bp[0]; b.u[1] = bp[1];
        acc[cb] = wmma16(a, b, acc[cb]);
      }
    }
    #pragma unroll
    for (int cb = 0; cb < 4; ++cb) {
      int col  = wn2 * 64 + cb * 16 + laneLo;
      float bs = bsi2[col];
      int rbase = wm2 * 16 + hiSel * 8;
      #pragma unroll
      for (int i = 0; i < 8; ++i)
        Atile[(rbase + i) * DIN + 512 + col] = (_Float16)(acc[cb][i] + bs);
    }
  }
  __syncthreads();

  // ---- Heads: 6 branches + speed head, A-tile reused for all 7 ----
  const int wm = wid >> 2;    // 0..1 -> rows wm*32
  const int wn = wid & 3;     // 0..3 -> cols wn*64

  for (int h = 0; h < NH; ++h) {
    // ---------------- Layer 1: [64,640] @ [640,256] ----------------
    {
      v8f acc[2][4];
      #pragma unroll
      for (int rb = 0; rb < 2; ++rb)
        #pragma unroll
        for (int cb = 0; cb < 4; ++cb) acc[rb][cb] = (v8f){0,0,0,0,0,0,0,0};

      const _Float16* W1h = W1s + (size_t)h * KS1 * HH * 32;
      for (int ks = 0; ks < KS1; ++ks) {
        int kh = ks * 32 + hiSel * 8;
        Frag a[2];
        #pragma unroll
        for (int rb = 0; rb < 2; ++rb) {
          int row = wm * 32 + rb * 16 + laneLo;
          a[rb].u[0] = *(const v4u*)&Atile[row * DIN + kh];
          a[rb].u[1] = *(const v4u*)&Atile[row * DIN + kh + 16];
        }
        if (ks + 1 < KS1)
          __builtin_prefetch(&W1h[((size_t)(ks + 1) * HH + wn * 64 + laneLo) * 32], 0, 0);
        #pragma unroll
        for (int cb = 0; cb < 4; ++cb) {
          int n = wn * 64 + cb * 16 + laneLo;
          const v4u* bp = (const v4u*)&W1h[((size_t)ks * HH + n) * 32 + hiSel * 16];
          Frag b; b.u[0] = bp[0]; b.u[1] = bp[1];
          acc[0][cb] = wmma16(a[0], b, acc[0][cb]);
          acc[1][cb] = wmma16(a[1], b, acc[1][cb]);
        }
      }
      const float* b1 = (h < 6) ? (bb1 + h * HH) : bso1;
      #pragma unroll
      for (int cb = 0; cb < 4; ++cb) {
        int col  = wn * 64 + cb * 16 + laneLo;
        float bs = b1[col];
        #pragma unroll
        for (int rb = 0; rb < 2; ++rb) {
          int rbase = wm * 32 + rb * 16 + hiSel * 8;
          #pragma unroll
          for (int i = 0; i < 8; ++i) {
            float v = acc[rb][cb][i] + bs;
            Xa[(rbase + i) * HH + col] = (_Float16)(v > 0.f ? v : 0.f);
          }
        }
      }
    }
    __syncthreads();

    // ---------------- Layer 2: [64,256] @ [256,256] ----------------
    {
      v8f acc[2][4];
      #pragma unroll
      for (int rb = 0; rb < 2; ++rb)
        #pragma unroll
        for (int cb = 0; cb < 4; ++cb) acc[rb][cb] = (v8f){0,0,0,0,0,0,0,0};

      const _Float16* W2h = W2s + (size_t)h * KS2 * HH * 32;
      for (int ks = 0; ks < KS2; ++ks) {
        int kh = ks * 32 + hiSel * 8;
        Frag a[2];
        #pragma unroll
        for (int rb = 0; rb < 2; ++rb) {
          int row = wm * 32 + rb * 16 + laneLo;
          a[rb].u[0] = *(const v4u*)&Xa[row * HH + kh];
          a[rb].u[1] = *(const v4u*)&Xa[row * HH + kh + 16];
        }
        #pragma unroll
        for (int cb = 0; cb < 4; ++cb) {
          int n = wn * 64 + cb * 16 + laneLo;
          const v4u* bp = (const v4u*)&W2h[((size_t)ks * HH + n) * 32 + hiSel * 16];
          Frag b; b.u[0] = bp[0]; b.u[1] = bp[1];
          acc[0][cb] = wmma16(a[0], b, acc[0][cb]);
          acc[1][cb] = wmma16(a[1], b, acc[1][cb]);
        }
      }
      const float* b2 = (h < 6) ? (bb2 + h * HH) : bso2;
      #pragma unroll
      for (int cb = 0; cb < 4; ++cb) {
        int col  = wn * 64 + cb * 16 + laneLo;
        float bs = b2[col];
        #pragma unroll
        for (int rb = 0; rb < 2; ++rb) {
          int rbase = wm * 32 + rb * 16 + hiSel * 8;
          #pragma unroll
          for (int i = 0; i < 8; ++i) {
            float v = acc[rb][cb][i] + bs;
            Xb[(rbase + i) * HH + col] = (_Float16)(v > 0.f ? v : 0.f);
          }
        }
      }
    }
    __syncthreads();

    // ---------------- Layer 3 weights -> LDS (f32, padded to 4 cols) ----
    for (int q = tid; q < HH * 4; q += 256) {
      int k = q >> 2, o = q & 3;
      float wv = 0.f;
      if (h < 6) { if (o < 3) wv = Wb3[((size_t)h * HH + k) * 3 + o]; }
      else if (o == 0) wv = Wso3[k];
      w3lds[q] = wv;
    }
    __syncthreads();

    // ---------------- Layer 3 dot + masked write ----------------
    {
      int row = tid >> 2, o = tid & 3;
      float a3 = (h < 6) ? ((o < 3) ? bb3[h * 3 + o] : 0.f)
                         : ((o == 0) ? bso3[0] : 0.f);
      for (int k = 0; k < HH; ++k)
        a3 = fmaf((float)Xb[row * HH + k], w3lds[k * 4 + o], a3);
      int rowG = mBase + row;
      if (h < 6) {
        if (o < 3 && cmd[rowG] == h + 1)
          outCtrl[(size_t)rowG * 3 + o] = 1.f / (1.f + __expf(-a3));
      } else if (o == 0) {
        outSpd[rowG] = a3;
      }
    }
    __syncthreads();
  }
}

// ------------------------------- launcher ---------------------------------
extern "C" void kernel_launch(void* const* d_in, const int* in_sizes, int n_in,
                              void* d_out, int out_size, void* d_ws, size_t ws_size,
                              hipStream_t stream) {
  (void)in_sizes; (void)n_in; (void)out_size; (void)ws_size;
  const float* emb   = (const float*)d_in[0];
  const float* speed = (const float*)d_in[1];
  const int*   cmd   = (const int*)d_in[2];
  const float* Wsi1  = (const float*)d_in[3];
  const float* bsi1  = (const float*)d_in[4];
  const float* Wsi2  = (const float*)d_in[5];
  const float* bsi2  = (const float*)d_in[6];
  const float* Wso1  = (const float*)d_in[7];
  const float* bso1  = (const float*)d_in[8];
  const float* Wso2  = (const float*)d_in[9];
  const float* bso2  = (const float*)d_in[10];
  const float* Wso3  = (const float*)d_in[11];
  const float* bso3  = (const float*)d_in[12];
  const float* Wb1   = (const float*)d_in[13];
  const float* bb1   = (const float*)d_in[14];
  const float* Wb2   = (const float*)d_in[15];
  const float* bb2   = (const float*)d_in[16];
  const float* Wb3   = (const float*)d_in[17];
  const float* bb3   = (const float*)d_in[18];

  _Float16* W1s   = (_Float16*)d_ws;                       // 7*20*256*32 halves
  _Float16* W2s   = W1s + (size_t)NH * KS1 * HH * 32;      // 7*8*256*32
  _Float16* Wsi2s = W2s + (size_t)NH * KS2 * HH * 32;      // 8*128*32

  float* outCtrl = (float*)d_out;
  float* outSpd  = outCtrl + (size_t)65536 * 3;

  const int totalPack = NH * DIN * HH + NH * HH * HH + HH * DLAT;
  cilrs_pack<<<(totalPack + 255) / 256, 256, 0, stream>>>(
      Wb1, Wso1, Wb2, Wso2, Wsi2, W1s, W2s, Wsi2s);

  cilrs_main<<<65536 / MT, 256, 0, stream>>>(
      emb, speed, cmd, Wsi1, bsi1, bsi2, bso1, bso2, bso3,
      bb1, bb2, bb3, Wso3, Wb3, W1s, W2s, Wsi2s, outCtrl, outSpd);
}
